// EMA_66769561584090
// MI455X (gfx1250) — matrix operand order repaired and checked by reading, not compiled, not run on hardware
//
#include <hip/hip_runtime.h>
#include <hip/hip_bf16.h>

typedef __attribute__((ext_vector_type(16))) __bf16 v16bf;
typedef __attribute__((ext_vector_type(8)))  __bf16 v8bf;
typedef __attribute__((ext_vector_type(4)))  __bf16 v4bf;
typedef __attribute__((ext_vector_type(8)))  float  v8f;

// Problem constants
#define DIMC   512
#define EDIMC  1024
#define OUTC   512
#define BC     4
#define LC     4096
#define MTOT   (BC*LC)       // 16384
#define CHUNK  128
#define NCHUNK (LC/CHUNK)    // 32

// ---------------------------------------------------------------------------
// p[e] = sigmoid(la[e]); q[e] = 1 - sigmoid(la)*sigmoid(ld)
// ---------------------------------------------------------------------------
__global__ void ema_pq_kernel(const float* __restrict__ la,
                              const float* __restrict__ ld,
                              float* __restrict__ p, float* __restrict__ q) {
    int e = blockIdx.x * blockDim.x + threadIdx.x;
    if (e < EDIMC) {
        float a = 1.0f / (1.0f + __expf(-la[e]));
        float d = 1.0f / (1.0f + __expf(-ld[e]));
        p[e] = a;
        q[e] = 1.0f - a * d;
    }
}

// ---------------------------------------------------------------------------
// bf16 WMMA GEMM:  D[M,N] = A[M,K] * B[K,N]
//   A: fp32 (converted in-register) or bf16; B: fp32 (converted); D: bf16 or f32
//   Block tile 128x128, K-step 32, 256 threads = 8 waves (2x4), wave tile 64x32
//   Software-pipelined: tile k+1 global loads issued while tile k WMMAs run.
// ---------------------------------------------------------------------------
template<bool A_IS_F32, bool OUT_IS_BF16>
__global__ __launch_bounds__(256)
void ema_wmma_gemm(const void* __restrict__ Aptr,
                   const float* __restrict__ Bptr,
                   void* __restrict__ Dptr,
                   int M, int N, int K) {
    constexpr int LDA = 40;                      // bf16 elems per LDS row (16B aligned, conflict-free)
    __shared__ __bf16 Alds[128 * LDA];
    __shared__ __bf16 Blds[128 * LDA];           // B stored transposed: Blds[n][k]

    const int tid  = threadIdx.x;
    const int lane = tid & 31;
    const int w    = tid >> 5;
    const int wm   = w & 1;                      // wave row (2)
    const int wn   = w >> 1;                     // wave col (4)
    const int g    = lane >> 4;                  // lane half
    const int hr   = lane & 15;                  // row/col within 16
    const int bm   = blockIdx.y * 128;
    const int bn   = blockIdx.x * 128;

    // ---- per-thread staging offsets (element units), kt added at load time ----
    const int acid = A_IS_F32 ? (tid & 7) : (tid & 3);
    const int arid = A_IS_F32 ? (tid >> 3) : (tid >> 2);
    const int bcid = tid & 31;
    const int brid = tid >> 5;
    size_t aoff[4];
    if constexpr (A_IS_F32) {
        #pragma unroll
        for (int it = 0; it < 4; ++it)
            aoff[it] = (size_t)(bm + arid + 32 * it) * K + acid * 4;
    } else {
        #pragma unroll
        for (int it = 0; it < 2; ++it)
            aoff[it] = (size_t)(bm + arid + 64 * it) * K + acid * 8;
    }
    size_t boff[4];
    #pragma unroll
    for (int it = 0; it < 4; ++it)
        boff[it] = (size_t)(brid + 8 * it) * N + bn + bcid * 4;

    // ---- staging registers ----
    float4 ar[4];   // A fp32 path
    v8bf   ah[2];   // A bf16 path
    float4 br[4];   // B fp32

    auto load_tile = [&](int kt) {
        if constexpr (A_IS_F32) {
            const float* A = (const float*)Aptr;
            #pragma unroll
            for (int it = 0; it < 4; ++it)
                ar[it] = *(const float4*)(A + aoff[it] + kt);
        } else {
            const __bf16* A = (const __bf16*)Aptr;
            #pragma unroll
            for (int it = 0; it < 2; ++it)
                ah[it] = *(const v8bf*)(A + aoff[it] + kt);
        }
        #pragma unroll
        for (int it = 0; it < 4; ++it)
            br[it] = *(const float4*)(Bptr + boff[it] + (size_t)kt * N);
    };

    auto store_tile = [&]() {
        if constexpr (A_IS_F32) {
            #pragma unroll
            for (int it = 0; it < 4; ++it) {
                v4bf h;
                h[0] = (__bf16)ar[it].x; h[1] = (__bf16)ar[it].y;
                h[2] = (__bf16)ar[it].z; h[3] = (__bf16)ar[it].w;
                *(v4bf*)(&Alds[(arid + 32 * it) * LDA + acid * 4]) = h;
            }
        } else {
            #pragma unroll
            for (int it = 0; it < 2; ++it)
                *(v8bf*)(&Alds[(arid + 64 * it) * LDA + acid * 8]) = ah[it];
        }
        #pragma unroll
        for (int it = 0; it < 4; ++it) {
            int kr = brid + 8 * it;
            Blds[(bcid * 4 + 0) * LDA + kr] = (__bf16)br[it].x;
            Blds[(bcid * 4 + 1) * LDA + kr] = (__bf16)br[it].y;
            Blds[(bcid * 4 + 2) * LDA + kr] = (__bf16)br[it].z;
            Blds[(bcid * 4 + 3) * LDA + kr] = (__bf16)br[it].w;
        }
    };

    v8f acc[4][2] = {};

    load_tile(0);                                 // prologue: tile 0 in flight

    for (int kt = 0; kt < K; kt += 32) {
        store_tile();                             // waits on in-flight loads here
        __syncthreads();

        if (kt + 32 < K)
            load_tile(kt + 32);                   // issue next tile; no wait — overlaps WMMAs

        // ---- load fragments from LDS ----
        v16bf afrag[4];
        #pragma unroll
        for (int mt = 0; mt < 4; ++mt) {
            int m = wm * 64 + mt * 16 + hr;
            v8bf lo = *(const v8bf*)(&Alds[m * LDA + 8 * g]);        // k = 8g .. 8g+7
            v8bf hi = *(const v8bf*)(&Alds[m * LDA + 16 + 8 * g]);   // k = 16+8g .. +7
            afrag[mt] = __builtin_shufflevector(lo, hi, 0,1,2,3,4,5,6,7,8,9,10,11,12,13,14,15);
        }
        v16bf bfrag[2];
        #pragma unroll
        for (int nt = 0; nt < 2; ++nt) {
            int n = wn * 32 + nt * 16 + hr;
            v8bf lo = *(const v8bf*)(&Blds[n * LDA + 8 * g]);        // k = 8g .. 8g+7
            v8bf hi = *(const v8bf*)(&Blds[n * LDA + 16 + 8 * g]);   // k = 16+8g .. +7
            bfrag[nt] = __builtin_shufflevector(lo, hi, 0,1,2,3,4,5,6,7,8,9,10,11,12,13,14,15);
        }

        // ---- 8 WMMAs per wave per k-step ----
        #pragma unroll
        for (int mt = 0; mt < 4; ++mt)
            #pragma unroll
            for (int nt = 0; nt < 2; ++nt)
                acc[mt][nt] = __builtin_amdgcn_wmma_f32_16x16x32_bf16(
                    false, afrag[mt], false, bfrag[nt],
                    (short)0, acc[mt][nt], false, false);

        __syncthreads();
    }

    // ---- epilogue: D element (m = r + 8g, n = hr) per VGPR r ----
    #pragma unroll
    for (int mt = 0; mt < 4; ++mt) {
        #pragma unroll
        for (int nt = 0; nt < 2; ++nt) {
            #pragma unroll
            for (int r = 0; r < 8; ++r) {
                int mg = bm + wm * 64 + mt * 16 + r + 8 * g;
                int ng = bn + wn * 32 + nt * 16 + hr;
                if constexpr (OUT_IS_BF16)
                    ((__bf16*)Dptr)[(size_t)mg * N + ng] = (__bf16)acc[mt][nt][r];
                else
                    ((float*)Dptr)[(size_t)mg * N + ng] = acc[mt][nt][r];
            }
        }
    }
}

// ---------------------------------------------------------------------------
// Chunked EMA scan over L (in place over the bf16 u buffer)
// ---------------------------------------------------------------------------
__global__ __launch_bounds__(256)
void ema_scan_local(__bf16* __restrict__ u, const float* __restrict__ p,
                    const float* __restrict__ q, float* __restrict__ csum) {
    int idx = blockIdx.x * blockDim.x + threadIdx.x;       // B*NCHUNK*EDIM threads
    int e = idx & (EDIMC - 1);
    int c = (idx >> 10) & (NCHUNK - 1);
    int b = idx >> 15;
    float pe = p[e], qe = q[e];
    size_t base = ((size_t)(b * LC + c * CHUNK)) * EDIMC + e;
    float y = 0.0f;
    #pragma unroll 4
    for (int t = 0; t < CHUNK; ++t) {
        size_t off = base + (size_t)t * EDIMC;
        float uv = (float)u[off];
        y = qe * y + pe * uv;
        u[off] = (__bf16)y;
    }
    csum[((size_t)(b * NCHUNK + c)) * EDIMC + e] = y;
}

__global__ __launch_bounds__(256)
void ema_scan_carry(const float* __restrict__ csum, const float* __restrict__ q,
                    float* __restrict__ carry) {
    int idx = blockIdx.x * blockDim.x + threadIdx.x;       // B*EDIM threads
    int e = idx & (EDIMC - 1);
    int b = idx >> 10;
    float qe = q[e];
    float Q = qe;
    #pragma unroll
    for (int i = 0; i < 7; ++i) Q *= Q;                    // q^128
    float st = 0.0f;
    for (int c = 0; c < NCHUNK; ++c) {
        size_t o = ((size_t)(b * NCHUNK + c)) * EDIMC + e;
        carry[o] = st;
        st = csum[o] + Q * st;
    }
}

__global__ __launch_bounds__(256)
void ema_scan_fix(__bf16* __restrict__ y, const float* __restrict__ q,
                  const float* __restrict__ carry) {
    int idx = blockIdx.x * blockDim.x + threadIdx.x;
    int e = idx & (EDIMC - 1);
    int c = (idx >> 10) & (NCHUNK - 1);
    int b = idx >> 15;
    float cr = carry[((size_t)(b * NCHUNK + c)) * EDIMC + e];
    float qe = q[e];
    float f = qe;                                          // q^(t+1)
    size_t base = ((size_t)(b * LC + c * CHUNK)) * EDIMC + e;
    #pragma unroll 4
    for (int t = 0; t < CHUNK; ++t) {
        size_t off = base + (size_t)t * EDIMC;
        y[off] = (__bf16)((float)y[off] + cr * f);
        f *= qe;
    }
}

// ---------------------------------------------------------------------------
extern "C" void kernel_launch(void* const* d_in, const int* in_sizes, int n_in,
                              void* d_out, int out_size, void* d_ws, size_t ws_size,
                              hipStream_t stream) {
    const float* emb  = (const float*)d_in[0];   // [B,L,DIM] fp32
    const float* la   = (const float*)d_in[1];   // [E]
    const float* ld   = (const float*)d_in[2];   // [E]
    const float* beta = (const float*)d_in[3];   // [DIM,E]
    const float* eta  = (const float*)d_in[4];   // [E,OUT]
    float* out = (float*)d_out;

    // workspace carve-up (~34.6 MB)
    char* ws = (char*)d_ws;
    __bf16* u    = (__bf16*)ws;                              // 16384*1024*2 = 32 MiB (u, then y in place)
    float* p     = (float*)(ws + (size_t)MTOT * EDIMC * 2);
    float* q     = p + EDIMC;
    float* csum  = q + EDIMC;                                // B*NCHUNK*E fp32 = 512 KiB
    float* carry = csum + (size_t)BC * NCHUNK * EDIMC;       // 512 KiB

    ema_pq_kernel<<<(EDIMC + 255) / 256, 256, 0, stream>>>(la, ld, p, q);

    // GEMM1: u[M,E] = emb[M,D] @ beta[D,E]  (f32 in -> bf16 out)
    ema_wmma_gemm<true, true><<<dim3(EDIMC / 128, MTOT / 128), 256, 0, stream>>>(
        (const void*)emb, beta, (void*)u, MTOT, EDIMC, DIMC);

    // chunked EMA scan, in place over u
    ema_scan_local<<<(BC * NCHUNK * EDIMC) / 256, 256, 0, stream>>>(u, p, q, csum);
    ema_scan_carry<<<(BC * EDIMC) / 256, 256, 0, stream>>>(csum, q, carry);
    ema_scan_fix  <<<(BC * NCHUNK * EDIMC) / 256, 256, 0, stream>>>(u, q, carry);

    // GEMM2: out[M,O] = y[M,E] @ eta[E,O]  (bf16 A -> f32 out)
    ema_wmma_gemm<false, false><<<dim3(OUTC / 128, MTOT / 128), 256, 0, stream>>>(
        (const void*)u, eta, (void*)out, MTOT, OUTC, EDIMC);
}
